// SimpleMambaLayer_12541304504372
// MI455X (gfx1250) — matrix operand order repaired and checked by reading, not compiled, not run on hardware
//
#include <hip/hip_runtime.h>
#include <stdint.h>

// ---------------------------------------------------------------------------
// SimpleMambaLayer for MI455X (gfx1250, wave32, WMMA)
//   xz = x W_in^T + b ; dt = sigmoid(x_part W_dt^T + b) ;
//   h_t = (1-dt) h + dt x_part ; y = (h * sigmoid(gate)) W_out^T + b
// Matmuls: v_wmma_f32_16x16x32_bf16. The shared 64x32 weight tile is staged
// into LDS with gfx1250 async-to-LDS loads (global_load_async_to_lds_b128,
// ASYNCcnt), double-buffered, one barrier per k-step. Scan is bandwidth-bound.
// ---------------------------------------------------------------------------

typedef __bf16 bf16_t;
typedef __attribute__((ext_vector_type(16))) __bf16 v16bf;
typedef __attribute__((ext_vector_type(8)))  float  v8f;

union Frag16 {
    v16bf v;
    uint4 q[2];
};

// -------------------------- fp32 -> bf16 downcast --------------------------
__global__ __launch_bounds__(256) void f32_to_bf16_kernel(
    const float* __restrict__ src, bf16_t* __restrict__ dst, long n)
{
    long i = (long)blockIdx.x * blockDim.x + threadIdx.x;
    long stride = (long)gridDim.x * blockDim.x;
    for (; i < n; i += stride) dst[i] = (bf16_t)src[i];
}

// ------------------------------ WMMA GEMM ----------------------------------
// C[M,N] = A[M,K](bf16 row-major) * W[N,K](bf16 row-major)^T + bias
// Block tile: 128(M) x 64(N); 8 waves, each owns a 16-row strip and shares a
// double-buffered 64x32 B tile in LDS, filled by async-to-LDS loads.
// K-step 32. Requires M%128==0, N%64==0, K%32==0 (holds: 32768/2048/1024).
//
// MODE 0: cols < Nhalf -> out0(f32) + out_bf(bf16) [x_part]; else out1 [gate]
// MODE 1: out0 = sigmoid(C)                        [dt]
// MODE 2: out0 = C                                 [y]
#define LDS_ROW   40                    // 32 elems + 8 pad (80B row stride)
#define BUF_BYTES (64 * LDS_ROW * 2)    // 5120 B per buffer

template <int MODE>
__global__ __launch_bounds__(256) void gemm_bf16_wmma(
    const bf16_t* __restrict__ A, const bf16_t* __restrict__ W,
    const float* __restrict__ bias, int M, int N, int K,
    float* __restrict__ out0, bf16_t* __restrict__ out_bf,
    float* __restrict__ out1, int Nhalf)
{
    __shared__ bf16_t ldsB[2][64 * LDS_ROW];        // 2 x 5 KB

    const int tid     = threadIdx.x;
    const int lane    = tid & 31;
    const int halfSel = lane >> 4;                  // 0: lanes 0-15, 1: 16-31
    const int lane16  = lane & 15;
    const int waveId  = tid >> 5;                   // 0..7

    const int nTiles = N >> 6;
    const long m0 = (long)(blockIdx.x / nTiles) * 128 + waveId * 16;
    const int  n0 = (int)(blockIdx.x % nTiles) * 64;

    // A fragment (16x32 bf16): lane holds row M = lane%16.
    // half 0: K in {k0+0..7, k0+16..23}; half 1: {k0+8..15, k0+24..31}.
    const bf16_t* Arow = A + (m0 + lane16) * (long)K + halfSel * 8;

    // B tile staging: thread t moves 8 contiguous bf16 of W row (n0 + t/4)
    // straight into LDS via the gfx1250 async-to-LDS path.
    const int sRow = tid >> 2;                      // 0..63
    const int sK   = (tid & 3) * 8;                 // 0,8,16,24
    const bf16_t* Wsrc = W + (long)(n0 + sRow) * K + sK;
    // LDS byte offset of this thread's 16B staging slot in buffer 0.
    const uint32_t sOff0 =
        (uint32_t)(uintptr_t)(&ldsB[0][0] + sRow * LDS_ROW + sK);

    const int KT = K >> 5;                          // k-steps

    // ---- prologue: async-stage B tile 0 into buf0, load A frag 0 ----
    {
        uint64_t ga = (uint64_t)(uintptr_t)Wsrc;
        asm volatile("global_load_async_to_lds_b128 %0, %1, off"
                     :: "v"(sOff0), "v"(ga) : "memory");
    }
    Frag16 aCur;
    aCur.q[0] = *reinterpret_cast<const uint4*>(Arow);
    aCur.q[1] = *reinterpret_cast<const uint4*>(Arow + 16);
    asm volatile("s_wait_asynccnt 0x0" ::: "memory");
    __syncthreads();

    v8f acc[4] = {};
    const bf16_t* bBase0 = &ldsB[0][0] + halfSel * 8;  // + row*LDS_ROW later

    for (int kt = 0; kt < KT; ++kt) {
        const int k0 = kt << 5;
        const bool more = (kt + 1) < KT;

        // ---- async-stage next B tile into the other LDS buffer.
        // Safe: every wave passed the previous barrier only after its WMMAs
        // consumed that buffer, so no wave is still reading it.
        Frag16 aNext;
        if (more) {
            uint64_t ga  = (uint64_t)(uintptr_t)(Wsrc + k0 + 32);
            uint32_t dst = sOff0 + (((kt + 1) & 1) ? (uint32_t)BUF_BYTES : 0u);
            asm volatile("global_load_async_to_lds_b128 %0, %1, off"
                         :: "v"(dst), "v"(ga) : "memory");
            aNext.q[0] = *reinterpret_cast<const uint4*>(Arow + k0 + 32);
            aNext.q[1] = *reinterpret_cast<const uint4*>(Arow + k0 + 48);
            __builtin_prefetch(Arow + k0 + 96, 0, 3);   // global_prefetch_b8
        }

        // ---- preload all 4 B fragments, then the WMMA chain (one dscnt wait
        // covers the group instead of one wait per WMMA).
        const bf16_t* bb = bBase0 + (kt & 1) * (64 * LDS_ROW);
        Frag16 bfr[4];
#pragma unroll
        for (int f = 0; f < 4; ++f) {
            const bf16_t* bp = bb + (f * 16 + lane16) * LDS_ROW;
            bfr[f].q[0] = *reinterpret_cast<const uint4*>(bp);
            bfr[f].q[1] = *reinterpret_cast<const uint4*>(bp + 16);
        }
#pragma unroll
        for (int f = 0; f < 4; ++f) {
            acc[f] = __builtin_amdgcn_wmma_f32_16x16x32_bf16(
                /*neg_a=*/false, aCur.v, /*neg_b=*/false, bfr[f].v,
                /*c_mod=*/(short)0, acc[f],
                /*reuse_a=*/false, /*reuse_b=*/false);
        }

        // ---- make next buffer visible to the whole block ----
        if (more) {
            asm volatile("s_wait_asynccnt 0x0" ::: "memory");
            __syncthreads();
            aCur = aNext;
        }
    }

    // ---- epilogue. D layout: VGPR r -> row = r + 8*halfSel, col = lane%16.
    const int rowHi = halfSel * 8;
#pragma unroll
    for (int f = 0; f < 4; ++f) {
        const int col = n0 + f * 16 + lane16;
        const float bv = bias[col];
#pragma unroll
        for (int r = 0; r < 8; ++r) {
            const long row = m0 + r + rowHi;
            float v = acc[f][r] + bv;
            if (MODE == 0) {
                if (col < Nhalf) {
                    long o = row * (long)Nhalf + col;
                    out0[o]   = v;
                    out_bf[o] = (bf16_t)v;
                } else {
                    out1[row * (long)Nhalf + (col - Nhalf)] = v;
                }
            } else if (MODE == 1) {
                out0[row * (long)N + col] = 1.0f / (1.0f + __expf(-v));
            } else {
                out0[row * (long)N + col] = v;
            }
        }
    }
}

// ------------------------------- scan --------------------------------------
// One thread per (b,d) channel; serial over T. h=(1-dt)h+dt*x = fma(dt,x-h,h).
// Emits s = h * sigmoid(gate) as bf16 (A operand of the final GEMM).
__global__ __launch_bounds__(256) void scan_kernel(
    const float* __restrict__ dt, const float* __restrict__ xp,
    const float* __restrict__ gate, const float* __restrict__ h0,
    bf16_t* __restrict__ s_bf, int B, int T, int D)
{
    long idx = (long)blockIdx.x * blockDim.x + threadIdx.x;
    if (idx >= (long)B * D) return;
    long b = idx / D, d = idx % D;
    float h = h0[idx];
    long base = b * (long)T * D + d;
    for (int t = 0; t < T; ++t, base += D) {
        float dtv = dt[base];
        float xv  = xp[base];
        h = fmaf(dtv, xv - h, h);
        float g = gate[base];
        float s = h * (1.0f / (1.0f + __expf(-g)));
        s_bf[base] = (bf16_t)s;
    }
}

// ------------------------------ launcher -----------------------------------
extern "C" void kernel_launch(void* const* d_in, const int* in_sizes, int n_in,
                              void* d_out, int out_size, void* d_ws, size_t ws_size,
                              hipStream_t stream)
{
    // setup_inputs order: x, h0, W_in, b_in, W_dt, b_dt, W_out, b_out
    const float* x     = (const float*)d_in[0];
    const float* h0    = (const float*)d_in[1];
    const float* W_in  = (const float*)d_in[2];
    const float* b_in  = (const float*)d_in[3];
    const float* W_dt  = (const float*)d_in[4];
    const float* b_dt  = (const float*)d_in[5];
    const float* W_out = (const float*)d_in[6];
    const float* b_out = (const float*)d_in[7];
    float* y = (float*)d_out;

    const int D   = in_sizes[5];              // 1024
    const int Bsz = in_sizes[1] / D;          // 16
    const int T   = in_sizes[0] / (Bsz * D);  // 2048
    const long M  = (long)Bsz * T;            // 32768
    const int N1  = 2 * D;                    // 2048

    // ---- workspace layout (~615 MB), 256B-aligned slices ----
    char* ws = (char*)d_ws;
    size_t off = 0;
    auto take = [&](size_t bytes) -> char* {
        char* p = ws + off;
        off = (off + bytes + 255) & ~(size_t)255;
        return p;
    };
    bf16_t* x_bf     = (bf16_t*)take(M * D * sizeof(bf16_t));          //  64 MB
    bf16_t* win_bf   = (bf16_t*)take((size_t)N1 * D * sizeof(bf16_t)); //   4 MB
    bf16_t* wdt_bf   = (bf16_t*)take((size_t)D * D * sizeof(bf16_t));  //   2 MB
    bf16_t* wout_bf  = (bf16_t*)take((size_t)D * D * sizeof(bf16_t));  //   2 MB
    float*  xpart_f  = (float*)take(M * D * sizeof(float));            // 128 MB
    float*  gate_f   = (float*)take(M * D * sizeof(float));            // 128 MB
    bf16_t* xpart_bf = (bf16_t*)take(M * D * sizeof(bf16_t));          //  64 MB
    float*  dt_f     = (float*)take(M * D * sizeof(float));            // 128 MB
    bf16_t* s_bf     = (bf16_t*)take(M * D * sizeof(bf16_t));          //  64 MB
    (void)ws_size; (void)n_in; (void)out_size;

    // ---- 1) downcast operands to bf16 ----
    f32_to_bf16_kernel<<<4096, 256, 0, stream>>>(x,     x_bf,   M * (long)D);
    f32_to_bf16_kernel<<<2048, 256, 0, stream>>>(W_in,  win_bf, (long)N1 * D);
    f32_to_bf16_kernel<<<1024, 256, 0, stream>>>(W_dt,  wdt_bf, (long)D * D);
    f32_to_bf16_kernel<<<1024, 256, 0, stream>>>(W_out, wout_bf,(long)D * D);

    // ---- 2) in_proj GEMM: xz -> x_part(f32+bf16), gate(f32) ----
    {
        unsigned blocks = (unsigned)((M / 128) * (N1 / 64));    // 8192
        gemm_bf16_wmma<0><<<blocks, 256, 0, stream>>>(
            x_bf, win_bf, b_in, (int)M, N1, D,
            xpart_f, xpart_bf, gate_f, D);
    }

    // ---- 3) dt GEMM with fused sigmoid ----
    {
        unsigned blocks = (unsigned)((M / 128) * (D / 64));     // 4096
        gemm_bf16_wmma<1><<<blocks, 256, 0, stream>>>(
            xpart_bf, wdt_bf, b_dt, (int)M, D, D,
            dt_f, nullptr, nullptr, 0);
    }

    // ---- 4) gated-EMA scan, fused gate sigmoid, bf16 output ----
    scan_kernel<<<dim3((Bsz * D + 255) / 256), 256, 0, stream>>>(
        dt_f, xpart_f, gate_f, h0, s_bf, Bsz, T, D);

    // ---- 5) out_proj GEMM -> y (f32) ----
    {
        unsigned blocks = (unsigned)((M / 128) * (D / 64));     // 4096
        gemm_bf16_wmma<2><<<blocks, 256, 0, stream>>>(
            s_bf, wout_bf, b_out, (int)M, D, D,
            y, nullptr, nullptr, 0);
    }
}
